// GINGraph_4217657884730
// MI455X (gfx1250) — compile-verified, matrix operand order checked by reading, not run on hardware
//
#include <hip/hip_runtime.h>

typedef __attribute__((ext_vector_type(16))) _Float16 v16h;
typedef __attribute__((ext_vector_type(8)))  _Float16 v8h;
typedef __attribute__((ext_vector_type(8)))  float    v8f;

#define N_NODES 100000
#define N_EDGES 1250000
#define DIM     64
#define HID     128
#define NGRAPH  2048
#define NLAYER  4

// ---------------------------------------------------------------- utilities
__global__ void k_zero(float* __restrict__ p, int n) {
  int i = blockIdx.x * 256 + threadIdx.x;
  if (i < n) p[i] = 0.0f;
}

// AtomEncoder: h[n,c] = sum_f atom_emb[f, x_feat[n,f], c]
__global__ void k_atom(const float* __restrict__ atom_emb,
                       const int* __restrict__ x_feat,
                       float* __restrict__ h) {
  int i = blockIdx.x * 256 + threadIdx.x;
  if (i >= N_NODES * DIM) return;
  int n = i >> 6, c = i & 63;
  float s = 0.f;
#pragma unroll
  for (int f = 0; f < 9; ++f) {
    int idx = x_feat[n * 9 + f];
    s += atom_emb[(f * 128 + idx) * DIM + c];
  }
  h[i] = s;
}

// Pack weights into WMMA-B-friendly f16 layout: per output column, K contiguous.
// W1 [L,64,128] -> W1t [L,128,64];  W2 [L,128,64] -> W2t [L,64,128]
__global__ void k_prep(const float* __restrict__ W1, const float* __restrict__ W2,
                       _Float16* __restrict__ W1t, _Float16* __restrict__ W2t) {
  int i = blockIdx.x * 256 + threadIdx.x;            // 0 .. 65535
  if (i < NLAYER * HID * DIM) {
    int k = i & 63, n = (i >> 6) & 127, l = i >> 13;
    W1t[(l * HID + n) * DIM + k] = (_Float16)W1[(l * DIM + k) * HID + n];
  } else {
    int j = i - NLAYER * HID * DIM;
    int k = j & 127, n = (j >> 7) & 63, l = j >> 13;
    W2t[(l * DIM + n) * HID + k] = (_Float16)W2[(l * HID + k) * DIM + n];
  }
}

// Per-edge message + scatter-add: 64 threads per edge (coalesced), f32 atomics hit L2.
__global__ void k_edge(const float* __restrict__ h, const float* __restrict__ bond_emb,
                       const int* __restrict__ edge_index, const int* __restrict__ edge_attr,
                       int layer, float* __restrict__ agg) {
  int t = blockIdx.x * 256 + threadIdx.x;
  int e = t >> 6;
  if (e >= N_EDGES) return;
  int c = t & 63;
  int src = edge_index[e];
  int dst = edge_index[N_EDGES + e];
  const float* bb = bond_emb + (size_t)(layer * 3) * 8 * DIM;
  float ev = bb[(0 * 8 + edge_attr[e * 3 + 0]) * DIM + c]
           + bb[(1 * 8 + edge_attr[e * 3 + 1]) * DIM + c]
           + bb[(2 * 8 + edge_attr[e * 3 + 2]) * DIM + c];
  float m = h[(size_t)src * DIM + c] + ev;
  m = m > 0.f ? m : 0.f;
  atomicAdd(&agg[(size_t)dst * DIM + c], m);
}

// GEMM1: u = z @ W1 + b1, z = (1+eps)*h + agg computed on the fly, staged f16 in LDS.
// Block = 256 thr = 8 waves; each wave owns one 16x16 tile of the 16x128 output stripe.
__global__ void __launch_bounds__(256)
k_gemm1(const float* __restrict__ h, const float* __restrict__ agg,
        const float* __restrict__ epsv, int layer,
        const _Float16* __restrict__ W1t, const float* __restrict__ b1,
        float* __restrict__ u) {
  __shared__ _Float16 zs[16 * 72];                   // 16 rows, padded stride 72 halves
  const int row0 = blockIdx.x * 16;
  const int t = threadIdx.x;
  {
    const float ep = 1.0f + epsv[layer];
    const int r = t >> 4;
    const int c4 = (t & 15) << 2;
    const float4 hv = *(const float4*)(h + (size_t)(row0 + r) * DIM + c4);
    const float4 av = *(const float4*)(agg + (size_t)(row0 + r) * DIM + c4);
    _Float16* zp = zs + r * 72 + c4;
    zp[0] = (_Float16)(ep * hv.x + av.x);
    zp[1] = (_Float16)(ep * hv.y + av.y);
    zp[2] = (_Float16)(ep * hv.z + av.z);
    zp[3] = (_Float16)(ep * hv.w + av.w);
  }
  __syncthreads();
  const int wave = t >> 5;                           // 0..7 (wave32)
  const int lane = t & 31;
  const int lrow = lane & 15;
  const int hi   = lane >> 4;                        // K-half select per ISA layout
  const int ch   = wave * 16 + lrow;                 // output channel (N dim)
  v8f acc = {};
  const _Float16* wb = W1t + (size_t)(layer * HID + ch) * DIM;
  const _Float16* za = zs + lrow * 72;
#pragma unroll
  for (int k0 = 0; k0 < DIM; k0 += 32) {
    v8h aLo = *(const v8h*)(za + k0 + hi * 8);
    v8h aHi = *(const v8h*)(za + k0 + 16 + hi * 8);
    v8h bLo = *(const v8h*)(wb + k0 + hi * 8);
    v8h bHi = *(const v8h*)(wb + k0 + 16 + hi * 8);
    v16h A = __builtin_shufflevector(aLo, aHi, 0,1,2,3,4,5,6,7,8,9,10,11,12,13,14,15);
    v16h B = __builtin_shufflevector(bLo, bHi, 0,1,2,3,4,5,6,7,8,9,10,11,12,13,14,15);
    acc = __builtin_amdgcn_wmma_f32_16x16x32_f16(false, A, false, B, (short)0, acc,
                                                 false, false);
  }
  const float bias = b1[layer * HID + ch];
#pragma unroll
  for (int i = 0; i < 8; ++i)
    u[(size_t)(row0 + hi * 8 + i) * HID + ch] = acc[i] + bias;
}

// Channel-wise sums for BatchNorm stats (thread = channel; coalesced row reads).
__global__ void k_bnstats(const float* __restrict__ x, int C, int rows_per_block,
                          float* __restrict__ sum, float* __restrict__ sumsq) {
  int c = threadIdx.x;
  int r0 = blockIdx.x * rows_per_block;
  int r1 = min(r0 + rows_per_block, N_NODES);
  float s = 0.f, q = 0.f;
  for (int r = r0; r < r1; ++r) {
    float v = x[(size_t)r * C + c];
    s += v; q += v * v;
  }
  atomicAdd(&sum[c], s);
  atomicAdd(&sumsq[c], q);
}

__global__ void k_bnfin(const float* __restrict__ sum, const float* __restrict__ sumsq,
                        const float* __restrict__ g, const float* __restrict__ b,
                        int layer, int C, float* __restrict__ scale,
                        float* __restrict__ shift) {
  int c = threadIdx.x;
  if (c >= C) return;
  float mean = sum[c] * (1.0f / N_NODES);
  float var  = sumsq[c] * (1.0f / N_NODES) - mean * mean;   // biased, as reference
  float sc = g[layer * C + c] * rsqrtf(var + 1e-5f);
  scale[c] = sc;
  shift[c] = b[layer * C + c] - mean * sc;
}

// GEMM2: v = relu(BN1(u)) @ W2 + b2, BN1+ReLU fused into the f16 A-fragment build.
// Block = 128 thr = 4 waves = 4 column tiles of the 64-wide output.
__global__ void __launch_bounds__(128)
k_gemm2(const float* __restrict__ u, const float* __restrict__ scale1,
        const float* __restrict__ shift1, int layer,
        const _Float16* __restrict__ W2t, const float* __restrict__ b2,
        float* __restrict__ v) {
  const int row0 = blockIdx.x * 16;
  const int t = threadIdx.x;
  const int wave = t >> 5;                           // 0..3
  const int lane = t & 31;
  const int lrow = lane & 15;
  const int hi   = lane >> 4;
  const int ch   = wave * 16 + lrow;
  const float* ub = u + (size_t)(row0 + lrow) * HID;
  const _Float16* wb = W2t + (size_t)(layer * DIM + ch) * HID;
  v8f acc = {};
#pragma unroll
  for (int k0 = 0; k0 < HID; k0 += 32) {
    const int ka = k0 + hi * 8;                      // first 8-half chunk
    const int kb = k0 + 16 + hi * 8;                 // second 8-half chunk
    v16h A;
#pragma unroll
    for (int j = 0; j < 16; ++j) {
      int k = (j < 8) ? (ka + j) : (kb + (j - 8));
      float y = ub[k] * scale1[k] + shift1[k];
      A[j] = (_Float16)(y > 0.f ? y : 0.f);
    }
    v8h bLo = *(const v8h*)(wb + ka);
    v8h bHi = *(const v8h*)(wb + kb);
    v16h B = __builtin_shufflevector(bLo, bHi, 0,1,2,3,4,5,6,7,8,9,10,11,12,13,14,15);
    acc = __builtin_amdgcn_wmma_f32_16x16x32_f16(false, A, false, B, (short)0, acc,
                                                 false, false);
  }
  const float bias = b2[layer * DIM + ch];
#pragma unroll
  for (int i = 0; i < 8; ++i)
    v[(size_t)(row0 + hi * 8 + i) * DIM + ch] = acc[i] + bias;
}

// Apply outer BN2 (+ReLU except last layer); destination is h or d_out (node repr).
__global__ void k_apply(const float* __restrict__ v, const float* __restrict__ scale2,
                        const float* __restrict__ shift2, int do_relu,
                        float* __restrict__ out) {
  int i = blockIdx.x * 256 + threadIdx.x;
  if (i >= N_NODES * DIM) return;
  int c = i & 63;
  float y = v[i] * scale2[c] + shift2[c];
  if (do_relu) y = fmaxf(y, 0.f);
  out[i] = y;
}

// Mean pooling by graph id.
__global__ void k_pool(const float* __restrict__ nr, const int* __restrict__ batch,
                       float* __restrict__ pool, float* __restrict__ cnt) {
  int i = blockIdx.x * 256 + threadIdx.x;
  if (i >= N_NODES * DIM) return;
  int n = i >> 6, c = i & 63;
  int g = batch[n];
  atomicAdd(&pool[(size_t)g * DIM + c], nr[i]);
  if (c == 0) atomicAdd(&cnt[g], 1.0f);
}

__global__ void k_pooldiv(const float* __restrict__ pool, const float* __restrict__ cnt,
                          float* __restrict__ out) {
  int i = blockIdx.x * 256 + threadIdx.x;
  if (i >= NGRAPH * DIM) return;
  out[i] = pool[i] / (cnt[i >> 6] + 1e-9f);
}

// ---------------------------------------------------------------- driver
extern "C" void kernel_launch(void* const* d_in, const int* in_sizes, int n_in,
                              void* d_out, int out_size, void* d_ws, size_t ws_size,
                              hipStream_t stream) {
  (void)in_sizes; (void)n_in; (void)out_size; (void)ws_size;
  const float* atom_emb = (const float*)d_in[0];
  const float* bond_emb = (const float*)d_in[1];
  const float* epsv     = (const float*)d_in[2];
  const float* W1       = (const float*)d_in[3];
  const float* b1       = (const float*)d_in[4];
  const float* bn1_g    = (const float*)d_in[5];
  const float* bn1_b    = (const float*)d_in[6];
  const float* W2       = (const float*)d_in[7];
  const float* b2       = (const float*)d_in[8];
  const float* bn2_g    = (const float*)d_in[9];
  const float* bn2_b    = (const float*)d_in[10];
  const int* x_feat     = (const int*)d_in[11];
  const int* edge_index = (const int*)d_in[12];
  const int* edge_attr  = (const int*)d_in[13];
  const int* batch      = (const int*)d_in[14];
  float* out = (float*)d_out;

  char* ws = (char*)d_ws;
  float*    h      = (float*)(ws);                       // N*64 f32
  float*    agg    = (float*)(ws + 25600000);            // N*64 f32 (reused as v)
  float*    u      = (float*)(ws + 51200000);            // N*128 f32
  _Float16* W1t    = (_Float16*)(ws + 102400000);        // 4*128*64 f16
  _Float16* W2t    = (_Float16*)(ws + 102465536);        // 4*64*128 f16
  float*    stats  = (float*)(ws + 102531072);           // 768 f32
  float*    sum1   = stats,        *sq1    = stats + 128;
  float*    scale1 = stats + 256,  *shift1 = stats + 384;
  float*    sum2   = stats + 512,  *sq2    = stats + 576;
  float*    scale2 = stats + 640,  *shift2 = stats + 704;
  float*    pool   = (float*)(ws + 102534144);           // B*64 f32
  float*    cnt    = pool + NGRAPH * DIM;                // B f32 (contiguous)
  float*    v      = agg;                                // alias: agg dead after GEMM1

  const int NT = N_NODES * DIM;                          // 6,400,000
  k_prep<<<256, 256, 0, stream>>>(W1, W2, W1t, W2t);
  k_atom<<<(NT + 255) / 256, 256, 0, stream>>>(atom_emb, x_feat, h);

  for (int l = 0; l < NLAYER; ++l) {
    k_zero<<<(NT + 255) / 256, 256, 0, stream>>>(agg, NT);
    k_zero<<<3, 256, 0, stream>>>(stats, 768);
    k_edge<<<(N_EDGES * DIM + 255) / 256, 256, 0, stream>>>(h, bond_emb, edge_index,
                                                            edge_attr, l, agg);
    k_gemm1<<<N_NODES / 16, 256, 0, stream>>>(h, agg, epsv, l, W1t, b1, u);
    k_bnstats<<<200, 128, 0, stream>>>(u, HID, 500, sum1, sq1);
    k_bnfin<<<1, 128, 0, stream>>>(sum1, sq1, bn1_g, bn1_b, l, HID, scale1, shift1);
    k_gemm2<<<N_NODES / 16, 128, 0, stream>>>(u, scale1, shift1, l, W2t, b2, v);
    k_bnstats<<<200, 64, 0, stream>>>(v, DIM, 500, sum2, sq2);
    k_bnfin<<<1, 64, 0, stream>>>(sum2, sq2, bn2_g, bn2_b, l, DIM, scale2, shift2);
    float* dst = (l == NLAYER - 1) ? out : h;
    k_apply<<<(NT + 255) / 256, 256, 0, stream>>>(v, scale2, shift2,
                                                  (l != NLAYER - 1) ? 1 : 0, dst);
  }
  k_zero<<<(NGRAPH * DIM + NGRAPH + 255) / 256, 256, 0, stream>>>(pool,
                                                                  NGRAPH * DIM + NGRAPH);
  k_pool<<<(NT + 255) / 256, 256, 0, stream>>>(out, batch, pool, cnt);
  k_pooldiv<<<(NGRAPH * DIM + 255) / 256, 256, 0, stream>>>(pool, cnt, out + NT);
}
